// PatchEmbeddingStage_3848290697754
// MI455X (gfx1250) — compile-verified, matrix operand order checked by reading, not run on hardware
//
#include <hip/hip_runtime.h>

typedef _Float16 v8h  __attribute__((ext_vector_type(8)));
typedef _Float16 v16h __attribute__((ext_vector_type(16)));
typedef float    v8f  __attribute__((ext_vector_type(8)));

union U16h { v16h v; v8h h[2]; };

static __device__ __forceinline__ v8f wmma_f16(v16h a, v16h b, v8f c) {
  // D = A(16x32 f16) * B(32x16 f16) + C(16x16 f32)
  return __builtin_amdgcn_wmma_f32_16x16x32_f16(
      /*neg_a=*/false, a, /*neg_b=*/false, b,
      /*c_mod=*/(short)0, c, /*reuse_a=*/false, /*reuse_b=*/false);
}

// ---------------------------------------------------------------------------
// Weight packing: fold BN scale, emit f16 in exact WMMA B-fragment lane order.
// B-fragment (32x16 f16): lane l -> n = l&15, khalf = l>>4;
// element j -> K = j + 8*khalf + (j>=8 ? 8 : 0).
// packed layout: [kt][nt(0..15)][lane(0..31)][j(0..15)] contiguous halves.
// K ordering: k = (kh*3+kw)*Cin + c
// ---------------------------------------------------------------------------
__global__ __launch_bounds__(256) void pack_kernel(
    const float* __restrict__ w, const float* __restrict__ gamma,
    const float* __restrict__ var, _Float16* __restrict__ dst,
    int Cin, int cshift, int KHW, int total)
{
  int flat = blockIdx.x * 256 + threadIdx.x;
  if (flat >= total) return;
  int j    = flat & 15;
  int lane = (flat >> 4) & 31;
  int nt   = (flat >> 9) & 15;
  int kt   = flat >> 13;
  int khalf = lane >> 4;
  int n     = lane & 15;
  int kf = j + (j >= 8 ? 8 : 0) + 8 * khalf;
  int k  = kt * 32 + kf;
  int e  = nt * 16 + n;
  int ck  = k & (Cin - 1);
  int khw = k >> cshift;
  float scale = gamma[e] * rsqrtf(var[e] + 1e-5f);
  dst[flat] = (_Float16)(w[(e * Cin + ck) * KHW + khw] * scale);
}

__global__ __launch_bounds__(256) void bias_kernel(
    const float* g3, const float* b3, const float* m3, const float* v3,
    const float* g4, const float* b4, const float* m4, const float* v4,
    const float* gr, const float* br, const float* mr, const float* vr,
    float* o3, float* o4, float* orr)
{
  int i = blockIdx.x * 256 + threadIdx.x;
  if (i < 256) {
    float sc = g3[i] * rsqrtf(v3[i] + 1e-5f);
    o3[i] = b3[i] - m3[i] * sc;
  } else if (i < 512) {
    int e = i - 256;
    float sc = g4[e] * rsqrtf(v4[e] + 1e-5f);
    o4[e] = b4[e] - m4[e] * sc;
  } else if (i < 768) {
    int e = i - 512;
    float sc = gr[e] * rsqrtf(vr[e] + 1e-5f);
    orr[e] = br[e] - mr[e] * sc;
  }
}

// ---------------------------------------------------------------------------
// Stage-1 LIF + layout transform:
//   x fp32 [T,B,C,H,W] -> spikes1 f16 [pixel][t][c], xh f16 same layout
//   pixel = (b*32 + h)*32 + w. One block per (b,h): C=128 x W=32 x T=4.
// ---------------------------------------------------------------------------
__global__ __launch_bounds__(256) void lif1_kernel(
    const float* __restrict__ x, _Float16* __restrict__ sp1,
    _Float16* __restrict__ xh)
{
  constexpr int PSTR = 132;              // padded row stride (halves)
  __shared__ _Float16 Ssp[32 * PSTR];
  __shared__ _Float16 Sx [32 * PSTR];
  const int tid = threadIdx.x;
  const int b = blockIdx.x >> 5, h = blockIdx.x & 31;
  float mem[16];
#pragma unroll
  for (int i = 0; i < 16; ++i) mem[i] = 0.f;

  for (int t4 = 0; t4 < 4; ++t4) {
#pragma unroll
    for (int i = 0; i < 16; ++i) {
      int e = i * 256 + tid;
      int c = e >> 5, w = e & 31;       // w fastest -> coalesced fp32 reads
      float xv = x[((t4 * 16 + b) * 128 + c) * 1024 + h * 32 + w];
      float m = mem[i] * 0.5f + xv;
      bool spk = (m - 1.0f) >= 0.f;
      Ssp[w * PSTR + c] = spk ? (_Float16)1.f : (_Float16)0.f;
      Sx [w * PSTR + c] = (_Float16)xv;
      mem[i] = spk ? 0.f : m;
    }
    __syncthreads();
#pragma unroll
    for (int i = 0; i < 16; ++i) {
      int e = i * 256 + tid;
      int w = e >> 7, c = e & 127;      // c fastest -> coalesced f16 writes
      int dst = (((b * 32 + h) * 32 + w) * 4 + t4) * 128 + c;
      sp1[dst] = Ssp[w * PSTR + c];
      xh [dst] = Sx [w * PSTR + c];
    }
    __syncthreads();
  }
}

// ---------------------------------------------------------------------------
// conv3 (3x3, C=128->E=256) + BN + LIF, implicit GEMM:
//   M = pixel*4 + t (64 rows/block = 16 w-pixels x 4 t), N = 256, K = 1152.
//   A staged in LDS (im2col with halo/zero-pad), B pre-packed per-lane.
//   Epilogue: bias + sequential LIF over the 4 t values held per lane.
// ---------------------------------------------------------------------------
__global__ __launch_bounds__(256) void conv3_kernel(
    const _Float16* __restrict__ sp1, const _Float16* __restrict__ pB,
    const float* __restrict__ bias, _Float16* __restrict__ sp2)
{
  constexpr int ASTR = 40;               // padded A row stride (halves)
  __shared__ _Float16 At[64 * ASTR];
  const int tid = threadIdx.x;
  const int wid = tid >> 5, lane = tid & 31;
  const int lrow = lane & 15, khalf = lane >> 4;
  const int pt = blockIdx.x;
  const int b = pt >> 6, rem = pt & 63, h = rem >> 1, w0 = (rem & 1) << 4;
  const int pixbase = (b * 32 + h) * 32 + w0;
  const int mrel = tid >> 2, q = tid & 3;
  const int pp = mrel >> 2, tt = mrel & 3;

  const v8f vzero = {0.f, 0.f, 0.f, 0.f, 0.f, 0.f, 0.f, 0.f};
  v8f acc[2][4];
#pragma unroll
  for (int s = 0; s < 2; ++s)
#pragma unroll
    for (int st = 0; st < 4; ++st) acc[s][st] = vzero;
  const v8h vz8 = {0.f, 0.f, 0.f, 0.f, 0.f, 0.f, 0.f, 0.f};

  for (int kt = 0; kt < 36; ++kt) {
    const int khw = kt >> 2, c0 = (kt & 3) << 5;
    const int dh = khw / 3 - 1, dw = khw - 3 * (khw / 3) - 1;
    const int hh = h + dh, ww = w0 + pp + dw;
    v8h val = vz8;
    if ((unsigned)hh < 32u && (unsigned)ww < 32u)
      val = *(const v8h*)(sp1 +
            ((((b * 32 + hh) * 32 + ww) * 4 + tt) * 128 + c0 + (q << 3)));
    *(v8h*)(At + mrel * ASTR + (q << 3)) = val;
    __syncthreads();

    U16h bf0, bf1;
    const _Float16* bp0 = pB + ((kt * 16 + wid * 2 + 0) * 512 + lane * 16);
    bf0.h[0] = *(const v8h*)(bp0);
    bf0.h[1] = *(const v8h*)(bp0 + 8);
    const _Float16* bp1 = pB + ((kt * 16 + wid * 2 + 1) * 512 + lane * 16);
    bf1.h[0] = *(const v8h*)(bp1);
    bf1.h[1] = *(const v8h*)(bp1 + 8);
    if (kt + 1 < 36)
      __builtin_prefetch(pB + (((kt + 1) * 16 + wid * 2) * 512 + lane * 16), 0, 0);

#pragma unroll
    for (int st = 0; st < 4; ++st) {
      const _Float16* ap = At + (st * 16 + lrow) * ASTR + (khalf << 3);
      U16h a;
      a.h[0] = *(const v8h*)(ap);        // K = 8*khalf + 0..7
      a.h[1] = *(const v8h*)(ap + 16);   // K = 16 + 8*khalf + 0..7
      acc[0][st] = wmma_f16(a.v, bf0.v, acc[0][st]);
      acc[1][st] = wmma_f16(a.v, bf1.v, acc[1][st]);
    }
    __syncthreads();
  }

  // Epilogue: bias + LIF (lane holds all 4 t per pixel group) -> f16 spikes
#pragma unroll
  for (int s = 0; s < 2; ++s) {
    const int ncol = wid * 32 + s * 16 + lrow;
    const float bb = bias[ncol];
#pragma unroll
    for (int st = 0; st < 4; ++st) {
#pragma unroll
      for (int g = 0; g < 2; ++g) {
        const int pl = st * 4 + g + 2 * khalf;
        _Float16* op = sp2 + ((pixbase + pl) * 4) * 256 + ncol;
        float mem = 0.f;
#pragma unroll
        for (int t4 = 0; t4 < 4; ++t4) {
          float v = acc[s][st][g * 4 + t4] + bb;
          mem = mem * 0.5f + v;
          const bool spk = (mem - 1.0f) >= 0.f;
          op[t4 * 256] = spk ? (_Float16)1.f : (_Float16)0.f;
          mem = spk ? 0.f : mem;
        }
      }
    }
  }
}

// ---------------------------------------------------------------------------
// conv4 (3x3, 256->256) + BN + LIF, then residual 1x1 GEMM over xh (+BN),
// then out = spike + residual, written NCHW fp32 via LDS transpose.
// ---------------------------------------------------------------------------
__global__ __launch_bounds__(256) void conv4_res_kernel(
    const _Float16* __restrict__ sp2, const _Float16* __restrict__ xh,
    const _Float16* __restrict__ pB4, const _Float16* __restrict__ pBr,
    const float* __restrict__ bias4, const float* __restrict__ biasr,
    float* __restrict__ out)
{
  constexpr int ASTR = 40;
  constexpr int OSTR = 257;              // padded fp32 out-tile stride
  __shared__ float     OutT[32 * OSTR];  // 32 M-rows x 256 e (half the tile)
  __shared__ _Float16  At[64 * ASTR];
  const int tid = threadIdx.x;
  const int wid = tid >> 5, lane = tid & 31;
  const int lrow = lane & 15, khalf = lane >> 4;
  const int pt = blockIdx.x;
  const int b = pt >> 6, rem = pt & 63, h = rem >> 1, w0 = (rem & 1) << 4;
  const int pixbase = (b * 32 + h) * 32 + w0;
  const int mrel = tid >> 2, q = tid & 3;
  const int pp = mrel >> 2, tt = mrel & 3;

  const v8f vzero = {0.f, 0.f, 0.f, 0.f, 0.f, 0.f, 0.f, 0.f};
  const v8h vz8 = {0.f, 0.f, 0.f, 0.f, 0.f, 0.f, 0.f, 0.f};
  v8f acc[2][4];
#pragma unroll
  for (int s = 0; s < 2; ++s)
#pragma unroll
    for (int st = 0; st < 4; ++st) acc[s][st] = vzero;

  // ---- conv4 GEMM: K = 9*256 = 2304, 72 k-steps ----
  for (int kt = 0; kt < 72; ++kt) {
    const int khw = kt >> 3, c0 = (kt & 7) << 5;
    const int dh = khw / 3 - 1, dw = khw - 3 * (khw / 3) - 1;
    const int hh = h + dh, ww = w0 + pp + dw;
    v8h val = vz8;
    if ((unsigned)hh < 32u && (unsigned)ww < 32u)
      val = *(const v8h*)(sp2 +
            ((((b * 32 + hh) * 32 + ww) * 4 + tt) * 256 + c0 + (q << 3)));
    *(v8h*)(At + mrel * ASTR + (q << 3)) = val;
    __syncthreads();

    U16h bf0, bf1;
    const _Float16* bp0 = pB4 + ((kt * 16 + wid * 2 + 0) * 512 + lane * 16);
    bf0.h[0] = *(const v8h*)(bp0);
    bf0.h[1] = *(const v8h*)(bp0 + 8);
    const _Float16* bp1 = pB4 + ((kt * 16 + wid * 2 + 1) * 512 + lane * 16);
    bf1.h[0] = *(const v8h*)(bp1);
    bf1.h[1] = *(const v8h*)(bp1 + 8);
    if (kt + 1 < 72)
      __builtin_prefetch(pB4 + (((kt + 1) * 16 + wid * 2) * 512 + lane * 16), 0, 0);

#pragma unroll
    for (int st = 0; st < 4; ++st) {
      const _Float16* ap = At + (st * 16 + lrow) * ASTR + (khalf << 3);
      U16h a;
      a.h[0] = *(const v8h*)(ap);
      a.h[1] = *(const v8h*)(ap + 16);
      acc[0][st] = wmma_f16(a.v, bf0.v, acc[0][st]);
      acc[1][st] = wmma_f16(a.v, bf1.v, acc[1][st]);
    }
    __syncthreads();
  }

  // ---- LIF: pack 64 spike bits per lane (bit = st*8 + g*4 + t) ----
  unsigned sb0 = 0u, sb1 = 0u;
#pragma unroll
  for (int s = 0; s < 2; ++s) {
    const int ncol = wid * 32 + s * 16 + lrow;
    const float bb = bias4[ncol];
    unsigned bits = 0u;
#pragma unroll
    for (int st = 0; st < 4; ++st) {
#pragma unroll
      for (int g = 0; g < 2; ++g) {
        float mem = 0.f;
#pragma unroll
        for (int t4 = 0; t4 < 4; ++t4) {
          float v = acc[s][st][g * 4 + t4] + bb;
          mem = mem * 0.5f + v;
          const bool spk = (mem - 1.0f) >= 0.f;
          if (spk) bits |= 1u << (st * 8 + g * 4 + t4);
          mem = spk ? 0.f : mem;
        }
      }
    }
    if (s == 0) sb0 = bits; else sb1 = bits;
  }

  // ---- residual 1x1 GEMM over raw-x f16: K = 128, 4 k-steps ----
#pragma unroll
  for (int s = 0; s < 2; ++s)
#pragma unroll
    for (int st = 0; st < 4; ++st) acc[s][st] = vzero;

  for (int kt = 0; kt < 4; ++kt) {
    const int c0 = kt << 5;
    v8h val = *(const v8h*)(xh +
              (((pixbase + pp) * 4 + tt) * 128 + c0 + (q << 3)));
    *(v8h*)(At + mrel * ASTR + (q << 3)) = val;
    __syncthreads();

    U16h bf0, bf1;
    const _Float16* bp0 = pBr + ((kt * 16 + wid * 2 + 0) * 512 + lane * 16);
    bf0.h[0] = *(const v8h*)(bp0);
    bf0.h[1] = *(const v8h*)(bp0 + 8);
    const _Float16* bp1 = pBr + ((kt * 16 + wid * 2 + 1) * 512 + lane * 16);
    bf1.h[0] = *(const v8h*)(bp1);
    bf1.h[1] = *(const v8h*)(bp1 + 8);

#pragma unroll
    for (int st = 0; st < 4; ++st) {
      const _Float16* ap = At + (st * 16 + lrow) * ASTR + (khalf << 3);
      U16h a;
      a.h[0] = *(const v8h*)(ap);
      a.h[1] = *(const v8h*)(ap + 16);
      acc[0][st] = wmma_f16(a.v, bf0.v, acc[0][st]);
      acc[1][st] = wmma_f16(a.v, bf1.v, acc[1][st]);
    }
    __syncthreads();
  }

  // ---- out = spike + residual + bias_r ; NCHW stores via LDS transpose ----
#pragma unroll
  for (int mc = 0; mc < 2; ++mc) {       // two 32-row chunks to bound LDS
    __syncthreads();
#pragma unroll
    for (int s = 0; s < 2; ++s) {
      const int ncol = wid * 32 + s * 16 + lrow;
      const float br = biasr[ncol];
      const unsigned sb = s ? sb1 : sb0;
#pragma unroll
      for (int stl = 0; stl < 2; ++stl) {
        const int st = mc * 2 + stl;
#pragma unroll
        for (int r = 0; r < 8; ++r) {
          const float spv = ((sb >> (st * 8 + r)) & 1u) ? 1.f : 0.f;
          OutT[(stl * 16 + r + 8 * khalf) * OSTR + ncol] =
              spv + acc[s][st][r] + br;
        }
      }
    }
    __syncthreads();
#pragma unroll 4
    for (int i = 0; i < 32; ++i) {
      const int idx = i * 256 + tid;
      const int e = idx >> 5, r2 = idx & 31, t4 = r2 >> 3, w8 = r2 & 7;
      out[(((t4 * 16 + b) * 256 + e) * 32 + h) * 32 + w0 + mc * 8 + w8] =
          OutT[(w8 * 4 + t4) * OSTR + e];
    }
  }
}

// ---------------------------------------------------------------------------
// Host-side orchestration (graph-capture safe: launches only).
// ---------------------------------------------------------------------------
extern "C" void kernel_launch(void* const* d_in, const int* in_sizes, int n_in,
                              void* d_out, int out_size, void* d_ws, size_t ws_size,
                              hipStream_t stream) {
  (void)in_sizes; (void)n_in; (void)out_size; (void)ws_size;
  const float* x  = (const float*)d_in[0];
  const float* w3 = (const float*)d_in[1];
  const float* g3 = (const float*)d_in[2];
  const float* b3 = (const float*)d_in[3];
  const float* m3 = (const float*)d_in[4];
  const float* v3 = (const float*)d_in[5];
  const float* w4 = (const float*)d_in[6];
  const float* g4 = (const float*)d_in[7];
  const float* b4 = (const float*)d_in[8];
  const float* m4 = (const float*)d_in[9];
  const float* v4 = (const float*)d_in[10];
  const float* wr = (const float*)d_in[11];
  const float* gr = (const float*)d_in[12];
  const float* br = (const float*)d_in[13];
  const float* mr = (const float*)d_in[14];
  const float* vr = (const float*)d_in[15];
  float* out = (float*)d_out;

  char* ws = (char*)d_ws;
  // workspace layout (all 256B aligned)
  _Float16* pB3   = (_Float16*)(ws + 0);                 // 1152*256*2 = 589,824
  _Float16* pB4   = (_Float16*)(ws + 589824);            // 2304*256*2 = 1,179,648
  _Float16* pBr   = (_Float16*)(ws + 1769472);           //  128*256*2 = 65,536
  float*    bb3   = (float*)   (ws + 1835008);           // 1024
  float*    bb4   = (float*)   (ws + 1836032);           // 1024
  float*    bbr   = (float*)   (ws + 1837056);           // 1024
  _Float16* sp1   = (_Float16*)(ws + 1838080);           // 16,777,216
  _Float16* xh    = (_Float16*)(ws + 18615296);          // 16,777,216
  _Float16* sp2   = (_Float16*)(ws + 35392512);          // 33,554,432
  // total ~68.9 MB

  // 1) fold BN + pre-swizzle weights into WMMA B-fragment order
  pack_kernel<<<1152, 256, 0, stream>>>(w3, g3, v3, pB3, 128, 7, 9, 1152 * 256);
  pack_kernel<<<2304, 256, 0, stream>>>(w4, g4, v4, pB4, 256, 8, 9, 2304 * 256);
  pack_kernel<<< 128, 256, 0, stream>>>(wr, gr, vr, pBr, 128, 7, 1,  128 * 256);
  bias_kernel<<<3, 256, 0, stream>>>(g3, b3, m3, v3, g4, b4, m4, v4,
                                     gr, br, mr, vr, bb3, bb4, bbr);
  // 2) stage-1 LIF + channels-last f16 transform (spikes + raw-x copy)
  lif1_kernel<<<512, 256, 0, stream>>>(x, sp1, xh);
  // 3) conv3 + BN + LIF (WMMA implicit GEMM, K=1152)
  conv3_kernel<<<1024, 256, 0, stream>>>(sp1, pB3, bb3, sp2);
  // 4) conv4 + BN + LIF + residual 1x1 + sum -> NCHW fp32 out
  conv4_res_kernel<<<1024, 256, 0, stream>>>(sp2, xh, pB4, pBr, bb4, bbr, out);
}
